// get_model_55121610277269
// MI455X (gfx1250) — matrix operand order repaired and checked by reading, not compiled, not run on hardware
//
#include <hip/hip_runtime.h>
#include <hip/hip_bf16.h>
#include <math.h>

#define NEGS 0.2f
#define RSC  0.9999950000374997f   /* 1/sqrt(1+1e-5) */

typedef __attribute__((ext_vector_type(16))) __bf16 v16bf;
typedef __attribute__((ext_vector_type(8)))  float  v8f;

__device__ __forceinline__ v8f wmma_bf16(v16bf a, v16bf b, v8f c) {
  return __builtin_amdgcn_wmma_f32_16x16x32_bf16(false, a, false, b, (short)0, c, false, false);
}

/* ------------------------------------------------------------------ */
/* Batched strided GEMM: Y = act(bn(W*X + bias)), bf16 WMMA 16x16x32  */
/* Wave computes 16 x (16*NT) tile. Loads are unconditional (clamped  */
/* index) + arithmetic masking -> no exec-mask branches in hot loop.  */
/* ------------------------------------------------------------------ */
struct GemmP {
  const float* A; long long aB1, aB2; int aR, aC;
  const float* X; long long xB1, xB2; int xR, xC;
  float*       Y; long long yB1, yB2; int yR, yC;
  int M, N, K, nB2;
  const float* bias; const float* g; const float* bt;
  int act;    // 0 none, 1 lrelu, 2 relu
  int accum;  // add into existing Y (after act)
};

template <int NT>
__global__ __launch_bounds__(32) void k_gemm(GemmP p) {
  int z = blockIdx.z; int b1 = z / p.nB2, b2 = z % p.nB2;
  const float* A = p.A + b1 * p.aB1 + b2 * p.aB2;
  const float* X = p.X + b1 * p.xB1 + b2 * p.xB2;
  float*       Y = p.Y + b1 * p.yB1 + b2 * p.yB2;
  int lane = threadIdx.x & 31, half = lane >> 4, l15 = lane & 15;
  int m0 = blockIdx.x * 16, n0 = blockIdx.y * 16 * NT;
  int row = m0 + l15;
  bool rok = row < p.M;
  const float* Arow = A + (rok ? row : p.M - 1) * p.aR;

  bool cok[NT]; const float* Xcol[NT];
  v8f acc[NT];
#pragma unroll
  for (int t = 0; t < NT; ++t) {
    int col = n0 + t * 16 + l15;
    cok[t] = col < p.N;
    Xcol[t] = X + (cok[t] ? col : p.N - 1) * p.xC;
    acc[t] = (v8f){0,0,0,0,0,0,0,0};
  }

  for (int k0 = 0; k0 < p.K; k0 += 32) {
    v16bf a;
#pragma unroll
    for (int j = 0; j < 16; ++j) {
      int v = j >> 1, h = j & 1;
      int kk = ((v < 4) ? (half*8 + 2*v + h) : (16 + half*8 + 2*(v-4) + h)) + k0;
      bool ok = rok && kk < p.K;
      float fa = Arow[(kk < p.K ? kk : p.K - 1) * p.aC];
      a[j] = (__bf16)(ok ? fa : 0.f);
    }
#pragma unroll
    for (int t = 0; t < NT; ++t) {
      v16bf bx;
#pragma unroll
      for (int j = 0; j < 16; ++j) {
        int k2 = k0 + ((j < 8) ? (half*8 + j) : (16 + half*8 + (j-8)));
        bool ok = cok[t] && k2 < p.K;
        float fx = Xcol[t][(k2 < p.K ? k2 : p.K - 1) * p.xR];
        bx[j] = (__bf16)(ok ? fx : 0.f);
      }
      acc[t] = wmma_bf16(a, bx, acc[t]);
    }
  }
#pragma unroll
  for (int t = 0; t < NT; ++t) {
    int col = n0 + t * 16 + l15;
#pragma unroll
    for (int r = 0; r < 8; ++r) {
      int rr = m0 + r + half*8;
      if (rr < p.M && col < p.N) {
        float y = acc[t][r];
        if (p.bias) y += p.bias[rr];
        if (p.g)    y = p.g[rr] * (y * RSC) + p.bt[rr];
        if (p.act == 1)      y = (y >= 0.f) ? y : NEGS * y;
        else if (p.act == 2) y = fmaxf(y, 0.f);
        float* py = &Y[rr * p.yR + col * p.yC];
        if (p.accum) y += *py;
        *py = y;
      }
    }
  }
}

/* ------------------------------------------------------------------ */
/* Flash attention from qkvT [batch, 3D, T] -> oT [batch, D, T]       */
/* sT = K*Q^T via WMMA, online softmax over keys, oT = V^T * pT       */
/* ------------------------------------------------------------------ */
struct FlashP {
  const float* qkv; long long bS;
  float* oT; long long oS;
  int T, D, H, hd; float scale;
};

__global__ void k_flash(FlashP p) {
  int b = blockIdx.z, h = blockIdx.y, qt = blockIdx.x;
  int wave = threadIdx.x >> 5, lane = threadIdx.x & 31;
  int half = lane >> 4, l15 = lane & 15;
  const float* base = p.qkv + (long long)b * p.bS;
  const float* Q  = base + h * p.hd * p.T;
  const float* Kv = base + (p.D + h * p.hd) * p.T;
  const float* Vv = base + (2 * p.D + h * p.hd) * p.T;
  int q0 = qt * 16, c0 = wave * 16;
  int qcol = q0 + l15;                         /* always < T */
  int cV = c0 + l15;
  const float* Vrow = Vv + (cV < p.hd ? cV : p.hd - 1) * p.T;
  bool vok = cV < p.hd;
  float m = -1e30f, l = 0.f;
  v8f o = {0,0,0,0,0,0,0,0};
  for (int kt = 0; kt < p.T; kt += 32) {
    v8f s0 = {0,0,0,0,0,0,0,0}, s1 = {0,0,0,0,0,0,0,0};
    int key0 = kt + l15, key1 = kt + 16 + l15;
    bool k0ok = key0 < p.T, k1ok = key1 < p.T;
    int key0c = k0ok ? key0 : p.T - 1, key1c = k1ok ? key1 : p.T - 1;
    for (int c = 0; c < p.hd; c += 32) {
      v16bf a0, a1, bq;
#pragma unroll
      for (int j = 0; j < 16; ++j) {
        int v = j >> 1, hh = j & 1;
        int kk = ((v < 4) ? (half*8 + 2*v + hh) : (16 + half*8 + 2*(v-4) + hh)) + c;
        int kkc = kk < p.hd ? kk : p.hd - 1;
        bool kok = kk < p.hd;
        float f0 = Kv[kkc * p.T + key0c];
        float f1 = Kv[kkc * p.T + key1c];
        a0[j] = (__bf16)((kok && k0ok) ? f0 : 0.f);
        a1[j] = (__bf16)((kok && k1ok) ? f1 : 0.f);
        int ck = c + ((j < 8) ? (half*8 + j) : (16 + half*8 + (j-8)));
        float fq = Q[(ck < p.hd ? ck : p.hd - 1) * p.T + qcol];
        bq[j] = (__bf16)(ck < p.hd ? fq : 0.f);
      }
      s0 = wmma_bf16(a0, bq, s0);
      s1 = wmma_bf16(a1, bq, s1);
    }
    float e0[8], e1[8]; float mloc = -1e30f;
#pragma unroll
    for (int r = 0; r < 8; ++r) {
      int k0i = kt + r + half*8, k1i = k0i + 16;
      float v0 = (k0i < p.T) ? s0[r] * p.scale : -1e30f;
      float v1 = (k1i < p.T) ? s1[r] * p.scale : -1e30f;
      e0[r] = v0; e1[r] = v1; mloc = fmaxf(mloc, fmaxf(v0, v1));
    }
    mloc = fmaxf(mloc, __shfl_xor(mloc, 16, 32));
    float mn = fmaxf(m, mloc);
    float corr = __expf(m - mn);
    float ls = 0.f;
#pragma unroll
    for (int r = 0; r < 8; ++r) { e0[r] = __expf(e0[r] - mn); e1[r] = __expf(e1[r] - mn); ls += e0[r] + e1[r]; }
    ls += __shfl_xor(ls, 16, 32);
    l = l * corr + ls; m = mn;
#pragma unroll
    for (int r = 0; r < 8; ++r) o[r] *= corr;
    float x0[8], x1[8];
#pragma unroll
    for (int r = 0; r < 8; ++r) { x0[r] = __shfl_xor(e0[r], 16, 32); x1[r] = __shfl_xor(e1[r], 16, 32); }
    v16bf pb, av;
#pragma unroll
    for (int j = 0; j < 16; ++j) {
      int r = j & 7; int ho = (j >> 3) & 1;
      float val = (ho != half) ? (half ? x1[r] : x0[r]) : (half ? e1[r] : e0[r]);
      pb[j] = (__bf16)val;
      int v = j >> 1, hh = j & 1;
      int kk = half*16 + j; (void)v; (void)hh;
      int key = kt + kk;
      float fv = Vrow[(key < p.T ? key : p.T - 1)];
      av[j] = (__bf16)((vok && key < p.T) ? fv : 0.f);
    }
    o = wmma_bf16(av, pb, o);
  }
  float inv = 1.f / l;
  float* O = p.oT + (long long)b * p.oS + h * p.hd * p.T;
#pragma unroll
  for (int r = 0; r < 8; ++r) {
    int cc = c0 + r + half*8;
    if (cc < p.hd) O[cc * p.T + qcol] = o[r] * inv;
  }
}

/* ------------------------- glue kernels --------------------------- */

__global__ void k_fxyz(float* geo, const float* xyz, const int* idx, int B, int N) {
  long long t = blockIdx.x * (long long)blockDim.x + threadIdx.x;
  long long tot = (long long)B * N * 16;
  if (t >= tot) return;
  int kk = (int)(t & 15); long long bn = t >> 4; int n = (int)(bn % N); int b = (int)(bn / N);
  const float* pc = xyz + ((long long)b * N + n) * 3;
  int j = idx[t];
  const float* pj = xyz + ((long long)b * N + j) * 3;
  float rx = pc[0]-pj[0], ry = pc[1]-pj[1], rz = pc[2]-pj[2];
  float dist = sqrtf(rx*rx + ry*ry + rz*rz + 1e-12f);
  long long cs = (long long)N * 16;
  long long base = (long long)b * 10 * cs + (long long)n * 16 + kk;
  geo[base]        = dist;
  geo[base + cs]   = rx; geo[base + 2*cs] = ry; geo[base + 3*cs] = rz;
  geo[base + 4*cs] = pc[0]; geo[base + 5*cs] = pc[1]; geo[base + 6*cs] = pc[2];
  geo[base + 7*cs] = pj[0]; geo[base + 8*cs] = pj[1]; geo[base + 9*cs] = pj[2];
}

__global__ void k_gather(float* out, long long outB, long long chS,
                         const float* feat, long long featB,
                         const int* idx, int B, int C, int N) {
  long long t = blockIdx.x * (long long)blockDim.x + threadIdx.x;
  long long tot = (long long)B * C * N * 16;
  if (t >= tot) return;
  int kk = (int)(t & 15); long long r = t >> 4;
  int n = (int)(r % N); r /= N; int c = (int)(r % C); int b = (int)(r / C);
  int j = idx[((long long)b * N + n) * 16 + kk];
  out[(long long)b * outB + (long long)c * chS + (long long)n * 16 + kk] =
      feat[(long long)b * featB + (long long)c * N + j];
}

__global__ void k_mean(float* m, const float* x, long long bS, long long cS, int cnt, int C) {
  int blk = blockIdx.x; int c = blk % C, b = blk / C;
  const float* pt = x + (long long)b * bS + (long long)c * cS;
  float s = 0.f;
  for (int i = threadIdx.x; i < cnt; i += blockDim.x) s += pt[i];
  __shared__ float sh[128];
  sh[threadIdx.x] = s; __syncthreads();
  for (int o = 64; o > 0; o >>= 1) { if ((int)threadIdx.x < o) sh[threadIdx.x] += sh[threadIdx.x + o]; __syncthreads(); }
  if (threadIdx.x == 0) m[blk] = sh[0] / (float)cnt;
}

__global__ void k_dense(float* out, const float* in, const float* W, const float* bias,
                        int B, int O, int I, int act) {
  int t = blockIdx.x * blockDim.x + threadIdx.x;
  if (t >= B * O) return;
  int o = t % O, b = t / O;
  float s = bias ? bias[o] : 0.f;
  for (int i = 0; i < I; ++i) s += in[(long long)b * I + i] * W[(long long)o * I + i];
  if (act == 3)      s = s / (1.f + __expf(-s)) ;          /* silu */
  else if (act == 4) s = 1.f / (1.f + __expf(-s));         /* sigmoid */
  out[t] = s;
}

__global__ void k_scale(float* x, long long bS, long long cS, const float* s, int B, int C) {
  long long t = blockIdx.x * (long long)blockDim.x + threadIdx.x;
  long long tot = (long long)B * C * cS;
  if (t >= tot) return;
  long long i = t % cS; long long r = t / cS;
  int c = (int)(r % C); int b = (int)(r / C);
  x[(long long)b * bS + (long long)c * cS + i] *= s[b * C + c];
}

__global__ void k_softmax16(float* x, long long rows) {
  long long r = blockIdx.x * (long long)blockDim.x + threadIdx.x;
  if (r >= rows) return;
  float* p = x + r * 16;
  float mx = p[0];
  for (int i = 1; i < 16; ++i) mx = fmaxf(mx, p[i]);
  float s = 0.f;
  for (int i = 0; i < 16; ++i) { float e = __expf(p[i] - mx); p[i] = e; s += e; }
  float inv = 1.f / s;
  for (int i = 0; i < 16; ++i) p[i] *= inv;
}

__global__ void k_wsum(float* agg, const float* x, const float* w, long long rows) {
  long long r = blockIdx.x * (long long)blockDim.x + threadIdx.x;
  if (r >= rows) return;
  const float* px = x + r * 16; const float* pw = w + r * 16;
  float s = 0.f;
  for (int i = 0; i < 16; ++i) s += px[i] * pw[i];
  agg[r] = s;
}

__global__ void k_knn(int* nn, const float* xyz, int B, int N) {
  int t = blockIdx.x * blockDim.x + threadIdx.x;
  if (t >= B * N) return;
  int i = t % N, b = t / N;
  const float* pi = xyz + ((long long)b * N + i) * 3;
  float bd[16]; int bi[16];
  for (int q = 0; q < 16; ++q) { bd[q] = 1e30f; bi[q] = 0; }
  for (int j = 0; j < N; ++j) {
    if (j == i) continue;
    const float* pj = xyz + ((long long)b * N + j) * 3;
    float dx = pi[0]-pj[0], dy = pi[1]-pj[1], dz = pi[2]-pj[2];
    float d2 = dx*dx + dy*dy + dz*dz;
    if (d2 < bd[15]) {
      int q = 15;
      while (q > 0 && bd[q-1] > d2) { bd[q] = bd[q-1]; bi[q] = bi[q-1]; --q; }
      bd[q] = d2; bi[q] = j;
    }
  }
  for (int q = 0; q < 16; ++q) nn[(long long)t * 16 + q] = bi[q];
}

__global__ void k_gcn_agg(float* out, const float* in, const int* nn, const float* bias,
                          int B, int C, int N, int relu) {
  long long t = blockIdx.x * (long long)blockDim.x + threadIdx.x;
  long long tot = (long long)B * C * N;
  if (t >= tot) return;
  int n = (int)(t % N); long long r = t / N;
  int c = (int)(r % C); int b = (int)(r / C);
  const float* base = in + ((long long)b * C + c) * N;
  float s = base[n];
  const int* id = nn + ((long long)b * N + n) * 16;
  for (int q = 0; q < 16; ++q) s += base[id[q]];
  s = s * (1.f / 17.f) + bias[c];
  if (relu) s = fmaxf(s, 0.f);
  out[t] = s;
}

__global__ void k_addin(float* dst, const float* src, long long n) {
  long long t = blockIdx.x * (long long)blockDim.x + threadIdx.x;
  if (t < n) dst[t] += src[t];
}

__global__ void k_lrelu(float* x, long long n) {
  long long t = blockIdx.x * (long long)blockDim.x + threadIdx.x;
  if (t < n) { float v = x[t]; x[t] = (v >= 0.f) ? v : NEGS * v; }
}

__global__ void k_maxgather(float* out, const float* feat, const int* sub,
                            int B, int C, int Nsrc, int Ndst) {
  long long t = blockIdx.x * (long long)blockDim.x + threadIdx.x;
  long long tot = (long long)B * C * Ndst;
  if (t >= tot) return;
  int n = (int)(t % Ndst); long long r = t / Ndst;
  int c = (int)(r % C); int b = (int)(r / C);
  const float* base = feat + ((long long)b * C + c) * Nsrc;
  const int* id = sub + ((long long)b * Ndst + n) * 16;
  float m = -1e30f;
  for (int q = 0; q < 16; ++q) m = fmaxf(m, base[id[q]]);
  out[t] = m;
}

__global__ void k_copych(float* dst, long long dstB, const float* src, int B, int C, int N) {
  long long t = blockIdx.x * (long long)blockDim.x + threadIdx.x;
  long long tot = (long long)B * C * N;
  if (t >= tot) return;
  int n = (int)(t % N); long long r = t / N;
  int c = (int)(r % C); int b = (int)(r / C);
  dst[(long long)b * dstB + (long long)c * N + n] = src[t];
}

__global__ void k_interp(float* dst, long long dstB, const float* src, const int* idx,
                         int B, int C, int Nsrc, int Ndst) {
  long long t = blockIdx.x * (long long)blockDim.x + threadIdx.x;
  long long tot = (long long)B * C * Ndst;
  if (t >= tot) return;
  int n = (int)(t % Ndst); long long r = t / Ndst;
  int c = (int)(r % C); int b = (int)(r / C);
  int j = idx[(long long)b * Ndst + n];
  dst[(long long)b * dstB + (long long)c * Ndst + n] =
      src[((long long)b * C + c) * Nsrc + j];
}

__global__ void k_logsoftmax(float* out, const float* logits, int B, int N, int C) {
  long long t = blockIdx.x * (long long)blockDim.x + threadIdx.x;
  if (t >= (long long)B * N) return;
  int n = (int)(t % N); int b = (int)(t / N);
  const float* p = logits + (long long)b * C * N + n;
  float mx = -1e30f;
  for (int c = 0; c < C; ++c) mx = fmaxf(mx, p[(long long)c * N]);
  float s = 0.f;
  for (int c = 0; c < C; ++c) s += __expf(p[(long long)c * N] - mx);
  float ls = logf(s);
  float* po = out + ((long long)b * N + n) * C;
  for (int c = 0; c < C; ++c) po[c] = p[(long long)c * N] - mx - ls;
}

/* ------------------------- host helpers --------------------------- */

static inline long long cdivll(long long a, long long b) { return (a + b - 1) / b; }

static void launch_gemm(hipStream_t st, const GemmP& p, int nB1) {
  if (p.N >= 64) {
    dim3 g((unsigned)((p.M + 15) / 16), (unsigned)((p.N + 63) / 64), (unsigned)(nB1 * p.nB2));
    k_gemm<4><<<g, 32, 0, st>>>(p);
  } else {
    dim3 g((unsigned)((p.M + 15) / 16), (unsigned)((p.N + 15) / 16), (unsigned)(nB1 * p.nB2));
    k_gemm<1><<<g, 32, 0, st>>>(p);
  }
}

static void gemm_cm(hipStream_t st, const float* W, const float* bias, const float* g,
                    const float* bt, int act, int accum, int Cout, int Cin, int S, int Bb,
                    const float* X, long long xB, int xR,
                    float* Y, long long yB, int yR) {
  GemmP p{};
  p.A = W; p.aB1 = 0; p.aB2 = 0; p.aR = Cin; p.aC = 1;
  p.X = X; p.xB1 = xB; p.xB2 = 0; p.xR = xR; p.xC = 1;
  p.Y = Y; p.yB1 = yB; p.yB2 = 0; p.yR = yR; p.yC = 1;
  p.M = Cout; p.N = S; p.K = Cin; p.nB2 = 1;
  p.bias = bias; p.g = g; p.bt = bt; p.act = act; p.accum = accum;
  launch_gemm(st, p, Bb);
}

static void flash(hipStream_t st, const float* qkv, long long bS, float* oT, long long oS,
                  int T, int D, int batch) {
  int hd = D / 4;
  FlashP p{qkv, bS, oT, oS, T, D, 4, hd, 1.f / sqrtf((float)hd)};
  int nw = (hd + 15) / 16;
  dim3 g((unsigned)(T / 16), 4, (unsigned)batch);
  k_flash<<<g, nw * 32, 0, st>>>(p);
}

#define L1D(kern, total, ...) do { long long _t = (total); \
  kern<<<(unsigned)cdivll(_t,256), 256, 0, st>>>(__VA_ARGS__); } while (0)

struct Arena {
  char* b; size_t cap, off;
  float* f(size_t n) { float* r = (float*)(b + off); off += (n * 4 + 255) & ~(size_t)255; return r; }
  int*   ii(size_t n){ int*   r = (int*)  (b + off); off += (n * 4 + 255) & ~(size_t)255; return r; }
};

struct PW { const float* p; long long off; const float* take(long long n) { const float* r = p + off; off += n; return r; } };
struct Cbn { const float *W, *b, *bt, *g; };
static Cbn cbn(PW& w, int ci, int co) {
  Cbn c; c.W = w.take((long long)co * ci); c.b = w.take(co); c.bt = w.take(co); c.g = w.take(co); return c;
}
struct Mha { const float *in_b, *in_w, *out_b, *out_w; };
static Mha mhap(PW& w, int D) {
  Mha m; m.in_b = w.take(3 * D); m.in_w = w.take((long long)3 * D * D);
  m.out_b = w.take(D); m.out_w = w.take((long long)D * D); return m;
}

struct EncP {
  const float* ap1fc; Cbn ap1m;
  const float* ap2fc; Cbn ap2m;
  Mha attn;
  const float *gW1, *gW2, *gb1, *gb2;
  Cbn bmlp1, bmlp2;
  const float *seW1, *seW2, *seb1, *seb2;
  Mha gattn;
  Cbn mlp1, mlp2, shc;
};

/* ------------------------------------------------------------------ */
extern "C" void kernel_launch(void* const* d_in, const int* in_sizes, int n_in,
                              void* d_out, int out_size, void* d_ws, size_t ws_size,
                              hipStream_t stream) {
  hipStream_t st = stream;
  const int B = 2, KN = 16, NC = 13;
  const int Ns[6] = {4096, 2048, 1024, 512, 256, 128};
  const int DOUT[5] = {16, 64, 128, 256, 512};

  const float* features = (const float*)d_in[0];
  const float* xyz[5]; const int* nei[5]; const int* sub[5]; const int* itp[5];
  for (int i = 0; i < 5; ++i) {
    xyz[i] = (const float*)d_in[1 + 4 * i];
    nei[i] = (const int*)  d_in[2 + 4 * i];
    sub[i] = (const int*)  d_in[3 + 4 * i];
    itp[i] = (const int*)  d_in[4 + 4 * i];
  }
  /* ---- params (jax tree_leaves order: sorted dict keys) ---- */
  PW w{(const float*)d_in[21], 0};
  Cbn decP[5];
  const int dd[5][2] = {{1536,512},{768,256},{384,128},{160,32},{64,32}};
  for (int j = 0; j < 5; ++j) decP[j] = cbn(w, dd[j][0], dd[j][1]);
  Cbn dec0P = cbn(w, 1024, 1024);
  EncP ep[5];
  int din = 8;
  for (int i = 0; i < 5; ++i) {
    int do_ = DOUT[i], d = do_ / 2, red = do_ / 8;
    EncP& e = ep[i];
    e.ap1fc = w.take((long long)9 * d * d); e.ap1m = cbn(w, 3 * d, d);
    e.ap2fc = w.take((long long)9 * d * d); e.ap2m = cbn(w, 3 * d, do_);
    e.attn = mhap(w, d);
    e.gW1 = w.take((long long)d * (d/2)); e.gW2 = w.take((long long)(d/2) * d);
    e.gb1 = w.take(d/2); e.gb2 = w.take(d);
    e.bmlp1 = cbn(w, 10, d); e.bmlp2 = cbn(w, d, d);
    e.seW1 = w.take((long long)red * d); e.seW2 = w.take((long long)d * red);
    e.seb1 = w.take(red); e.seb2 = w.take(d);
    e.gattn = mhap(w, 2 * do_);
    e.mlp1 = cbn(w, din, d); e.mlp2 = cbn(w, do_, 2 * do_); e.shc = cbn(w, din, 2 * do_);
    din = 2 * do_;
  }
  Cbn fc0 = cbn(w, 6, 8), fc1 = cbn(w, 32, 64), fc2 = cbn(w, 64, 32);
  const float* fc3W = w.take(13 * 32); const float* fc3b = w.take(13);

  /* ---- workspace arena: persistent first ---- */
  Arena ar{(char*)d_ws, ws_size, 0};
  float* f0 = ar.f((size_t)B * 8 * Ns[0]);
  float* e0buf = ar.f((size_t)B * 32 * Ns[0]);
  const int C2s[5] = {32, 128, 256, 512, 1024};
  float* sbuf[5];
  for (int i = 0; i < 5; ++i) sbuf[i] = ar.f((size_t)B * C2s[i] * Ns[i + 1]);
  size_t scratch0 = ar.off;

  /* ---- fc0 ---- */
  gemm_cm(st, fc0.W, fc0.b, fc0.g, fc0.bt, 1, 0, 8, 6, Ns[0], B,
          features, (long long)6 * Ns[0], Ns[0], f0, (long long)8 * Ns[0], Ns[0]);

  const float* fin = f0; int Cin = 8;

  /* ---- encoder ---- */
  for (int i = 0; i < 5; ++i) {
    Arena sc{(char*)d_ws, ws_size, scratch0};
    const EncP& e = ep[i];
    int N = Ns[i], Np = Ns[i + 1], do_ = DOUT[i], d = do_ / 2, red = do_ / 8, C2 = 2 * do_;
    int NK = N * KN;
    long long catB = (long long)3 * d * NK;

    /* f1 = cbl(mlp1, fin) */
    float* f1 = sc.f((size_t)B * d * N);
    gemm_cm(st, e.mlp1.W, e.mlp1.b, e.mlp1.g, e.mlp1.bt, 1, 0, d, Cin, N, B,
            fin, (long long)Cin * N, N, f1, (long long)d * N, N);

    /* geometric features -> cat1 slot1 */
    float* geo = sc.f((size_t)B * 10 * NK);
    L1D(k_fxyz, (long long)B * N * KN, geo, xyz[i], nei[i], B, N);
    float* cat1 = sc.f((size_t)B * 3 * d * NK);
    gemm_cm(st, e.bmlp1.W, e.bmlp1.b, e.bmlp1.g, e.bmlp1.bt, 1, 0, d, 10, NK, B,
            geo, (long long)10 * NK, NK, cat1 + (long long)d * NK, catB, NK);

    float* qkvT = sc.f((size_t)B * N * 3 * d * KN);
    float* oT   = sc.f((size_t)B * N * d * KN);
    float* sem = sc.f((size_t)B * d);
    float* se1 = sc.f((size_t)B * red);
    float* se2 = sc.f((size_t)B * d);
    float* att = sc.f((size_t)B * 3 * d * NK);
    float* agg = sc.f((size_t)B * 3 * d * N);

    auto neigh_branch = [&](const float* feat, float* cat) {
      /* gather to cat slot0 */
      L1D(k_gather, (long long)B * d * N * KN, cat, catB, (long long)NK, feat, (long long)d * N, nei[i], B, d, N);
      /* SE block (in place on cat slot0) */
      k_mean<<<dim3((unsigned)(B * d)), 128, 0, st>>>(sem, cat, catB, (long long)NK, NK, d);
      L1D(k_dense, (long long)B * red, se1, sem, e.seW1, e.seb1, B, red, d, 3);
      L1D(k_dense, (long long)B * d,   se2, se1, e.seW2, e.seb2, B, d, red, 4);
      L1D(k_scale, (long long)B * d * NK, cat, catB, (long long)NK, se2, B, d);
      /* qkvT[bn,3d,16] */
      GemmP q{};
      q.A = e.attn.in_w; q.aB1 = 0; q.aB2 = 0; q.aR = d; q.aC = 1;
      q.X = cat; q.xB1 = catB; q.xB2 = KN; q.xR = NK; q.xC = 1;
      q.Y = qkvT; q.yB1 = (long long)N * 3 * d * KN; q.yB2 = (long long)3 * d * KN; q.yR = KN; q.yC = 1;
      q.M = 3 * d; q.N = KN; q.K = d; q.nB2 = N;
      q.bias = e.attn.in_b; q.g = nullptr; q.bt = nullptr; q.act = 0; q.accum = 0;
      launch_gemm(st, q, B);
      flash(st, qkvT, (long long)3 * d * KN, oT, (long long)d * KN, KN, d, B * N);
      /* out proj -> cat slot2 (raw [B*N,k,d] token layout) */
      GemmP o{};
      o.A = e.attn.out_w; o.aB1 = 0; o.aB2 = 0; o.aR = d; o.aC = 1;
      o.X = oT; o.xB1 = (long long)N * d * KN; o.xB2 = (long long)d * KN; o.xR = KN; o.xC = 1;
      o.Y = cat + (long long)2 * d * NK; o.yB1 = catB; o.yB2 = (long long)KN * d; o.yR = 1; o.yC = d;
      o.M = d; o.N = KN; o.K = d; o.nB2 = N;
      o.bias = e.attn.out_b; o.g = nullptr; o.bt = nullptr; o.act = 0; o.accum = 0;
      launch_gemm(st, o, B);
    };

    auto att_pool = [&](const float* cat, const float* fcw, const Cbn& mlp, int cout, float* out) {
      gemm_cm(st, fcw, nullptr, nullptr, nullptr, 0, 0, 3 * d, 3 * d, NK, B,
              cat, catB, NK, att, catB, NK);
      L1D(k_softmax16, (long long)B * 3 * d * N, att, (long long)B * 3 * d * N);
      L1D(k_wsum, (long long)B * 3 * d * N, agg, cat, att, (long long)B * 3 * d * N);
      gemm_cm(st, mlp.W, mlp.b, mlp.g, mlp.bt, 1, 0, cout, 3 * d, N, B,
              agg, (long long)3 * d * N, N, out, (long long)cout * N, N);
    };

    /* branch 1 + attentive pool 1 */
    neigh_branch(f1, cat1);
    float* fag = sc.f((size_t)B * d * N);
    att_pool(cat1, e.ap1fc, e.ap1m, d, fag);

    /* GCN on kNN graph (constant degree k+1 = 17) */
    int* nn = sc.ii((size_t)B * N * KN);
    L1D(k_knn, (long long)B * N, nn, xyz[i], B, N);
    int dh = d / 2;
    float* h1 = sc.f((size_t)B * dh * N);
    {
      GemmP p{};
      p.A = e.gW1; p.aB1 = 0; p.aB2 = 0; p.aR = 1; p.aC = dh;   /* W1^T */
      p.X = fag; p.xB1 = (long long)d * N; p.xB2 = 0; p.xR = N; p.xC = 1;
      p.Y = h1; p.yB1 = (long long)dh * N; p.yB2 = 0; p.yR = N; p.yC = 1;
      p.M = dh; p.N = N; p.K = d; p.nB2 = 1;
      p.bias = nullptr; p.g = nullptr; p.bt = nullptr; p.act = 0; p.accum = 0;
      launch_gemm(st, p, B);
    }
    float* h1a = sc.f((size_t)B * dh * N);
    L1D(k_gcn_agg, (long long)B * dh * N, h1a, h1, nn, e.gb1, B, dh, N, 1);
    float* h2 = sc.f((size_t)B * d * N);
    {
      GemmP p{};
      p.A = e.gW2; p.aB1 = 0; p.aB2 = 0; p.aR = 1; p.aC = d;    /* W2^T */
      p.X = h1a; p.xB1 = (long long)dh * N; p.xB2 = 0; p.xR = N; p.xC = 1;
      p.Y = h2; p.yB1 = (long long)d * N; p.yB2 = 0; p.yR = N; p.yC = 1;
      p.M = d; p.N = N; p.K = dh; p.nB2 = 1;
      p.bias = nullptr; p.g = nullptr; p.bt = nullptr; p.act = 0; p.accum = 0;
      launch_gemm(st, p, B);
    }
    float* gout = sc.f((size_t)B * d * N);
    L1D(k_gcn_agg, (long long)B * d * N, gout, h2, nn, e.gb2, B, d, N, 0);
    L1D(k_addin, (long long)B * d * N, fag, gout, (long long)B * d * N);

    /* cat2: f_xyz2 slot1, branch 2 slots 0/2, attentive pool 2 */
    float* cat2 = sc.f((size_t)B * 3 * d * NK);
    gemm_cm(st, e.bmlp2.W, e.bmlp2.b, e.bmlp2.g, e.bmlp2.bt, 1, 0, d, d, NK, B,
            cat1 + (long long)d * NK, catB, NK, cat2 + (long long)d * NK, catB, NK);
    neigh_branch(fag, cat2);
    float* fbb = sc.f((size_t)B * do_ * N);
    att_pool(cat2, e.ap2fc, e.ap2m, do_, fbb);

    /* residual head: lrelu(bn(mlp2(fbb)) + bn(shortcut(fin))) */
    float* f2 = sc.f((size_t)B * C2 * N);
    gemm_cm(st, e.mlp2.W, e.mlp2.b, e.mlp2.g, e.mlp2.bt, 0, 0, C2, do_, N, B,
            fbb, (long long)do_ * N, N, f2, (long long)C2 * N, N);
    gemm_cm(st, e.shc.W, e.shc.b, e.shc.g, e.shc.bt, 0, 1, C2, Cin, N, B,
            fin, (long long)Cin * N, N, f2, (long long)C2 * N, N);
    L1D(k_lrelu, (long long)B * C2 * N, f2, (long long)B * C2 * N);

    /* global attention */
    float* qg = sc.f((size_t)B * 3 * C2 * N);
    gemm_cm(st, e.gattn.in_w, e.gattn.in_b, nullptr, nullptr, 0, 0, 3 * C2, C2, N, B,
            f2, (long long)C2 * N, N, qg, (long long)3 * C2 * N, N);
    float* og = sc.f((size_t)B * C2 * N);
    flash(st, qg, (long long)3 * C2 * N, og, (long long)C2 * N, N, C2, B);
    float* eout = (i == 0) ? e0buf : sc.f((size_t)B * C2 * N);
    gemm_cm(st, e.gattn.out_w, e.gattn.out_b, nullptr, nullptr, 0, 0, C2, C2, N, B,
            og, (long long)C2 * N, N, eout, (long long)C2 * N, N);

    /* random sample (max over gathered neighbours) */
    L1D(k_maxgather, (long long)B * C2 * Np, sbuf[i], eout, sub[i], B, C2, N, Np);
    fin = sbuf[i]; Cin = C2;
  }

  /* ---- decoder ---- */
  Arena dz{(char*)d_ws, ws_size, scratch0};
  float* f = dz.f((size_t)B * 1024 * Ns[5]);
  gemm_cm(st, dec0P.W, dec0P.b, dec0P.g, dec0P.bt, 1, 0, 1024, 1024, Ns[5], B,
          sbuf[4], (long long)1024 * Ns[5], Ns[5], f, (long long)1024 * Ns[5], Ns[5]);
  int fC = 1024;
  const int decN[5] = {256, 512, 1024, 2048, 4096};
  const int encC[5] = {512, 256, 128, 32, 32};
  const float* encBuf[5] = {sbuf[3], sbuf[2], sbuf[1], sbuf[0], e0buf};
  for (int j = 0; j < 5; ++j) {
    int Nj = decN[j], Cen = encC[j], Cc = Cen + fC, Cout = dd[j][1];
    int Nsrc = Ns[5 - j];
    float* cat = dz.f((size_t)B * Cc * Nj);
    L1D(k_copych, (long long)B * Cen * Nj, cat, (long long)Cc * Nj, encBuf[j], B, Cen, Nj);
    L1D(k_interp, (long long)B * fC * Nj, cat + (long long)Cen * Nj, (long long)Cc * Nj,
        f, itp[4 - j], B, fC, Nsrc, Nj);
    float* fn_ = dz.f((size_t)B * Cout * Nj);
    gemm_cm(st, decP[j].W, decP[j].b, decP[j].g, decP[j].bt, 1, 0, Cout, Cc, Nj, B,
            cat, (long long)Cc * Nj, Nj, fn_, (long long)Cout * Nj, Nj);
    f = fn_; fC = Cout;
  }

  /* ---- head ---- */
  int N0 = Ns[0];
  float* t64 = dz.f((size_t)B * 64 * N0);
  gemm_cm(st, fc1.W, fc1.b, fc1.g, fc1.bt, 1, 0, 64, 32, N0, B,
          f, (long long)32 * N0, N0, t64, (long long)64 * N0, N0);
  float* t32 = dz.f((size_t)B * 32 * N0);
  gemm_cm(st, fc2.W, fc2.b, fc2.g, fc2.bt, 1, 0, 32, 64, N0, B,
          t64, (long long)64 * N0, N0, t32, (long long)32 * N0, N0);
  float* logits = dz.f((size_t)B * NC * N0);
  gemm_cm(st, fc3W, fc3b, nullptr, nullptr, 0, 0, NC, 32, N0, B,
          t32, (long long)32 * N0, N0, logits, (long long)NC * N0, N0);
  L1D(k_logsoftmax, (long long)B * N0, (float*)d_out, logits, B, N0, NC);

  (void)n_in; (void)in_sizes; (void)out_size;
}
